// OptLocalAttentionBlock_50113678409897
// MI455X (gfx1250) — matrix operand order repaired and checked by reading, not compiled, not run on hardware
//
#include <hip/hip_runtime.h>

#define B_      2
#define S_      2048
#define WIDTH_  2560
#define NH_     10
#define HD_     256
#define WINDOW_ 1024
#define SP_     (S_ + 32)      // padded KV length so 32-wide key blocks never read OOB

typedef __attribute__((ext_vector_type(16))) __bf16       v16bf;
typedef __attribute__((ext_vector_type(8)))  float        v8f;
typedef __attribute__((ext_vector_type(4)))  unsigned int u32x4;

union FragU { v16bf bf; u32x4 q[2]; };

__device__ __forceinline__ v16bf ld_frag2(const unsigned short* p0, const unsigned short* p1) {
  FragU f;
  f.q[0] = *reinterpret_cast<const u32x4*>(p0);
  f.q[1] = *reinterpret_cast<const u32x4*>(p1);
  return f.bf;
}

__device__ __forceinline__ unsigned short f2bf(float x) {   // f32 -> bf16 RNE
  unsigned int u = __float_as_uint(x);
  u += 0x7fffu + ((u >> 16) & 1u);
  return (unsigned short)(u >> 16);
}

// A fragment (16x32 bf16): lane r=lane&15 holds row r; hi half picks K chunks
// elements 0..7  = K[k0 + hi*8 .. +7], elements 8..15 = K[k0+16+hi*8 .. +7]
__device__ __forceinline__ v16bf frag_a(const unsigned short* X, int ld, int k0, int lane) {
  int r = lane & 15, hi = lane >> 4;
  const unsigned short* p = X + (size_t)r * ld + k0 + hi * 8;
  return ld_frag2(p, p + 16);
}

// B fragment (32x16 bf16): lane holds column n0+(lane&15); elements 0..7 = K[k0+hi*8 .. +7], 8..15 = K[k0+16+hi*8 .. +7]
// B[k][n] = W[n][k] with W row-major (N,K)  -> two contiguous 16B runs per lane
__device__ __forceinline__ v16bf frag_b(const unsigned short* W, int ld, int n0, int k0, int lane) {
  int c = lane & 15, hi = lane >> 4;
  const unsigned short* p = W + (size_t)(n0 + c) * ld + k0 + hi * 8;
  return ld_frag2(p, p + 16);
}

__device__ __forceinline__ v8f wmma_bf16(v16bf a, v16bf b, v8f c) {
  return __builtin_amdgcn_wmma_f32_16x16x32_bf16(false, a, false, b, (short)0, c, false, false);
}

__device__ __forceinline__ float half_max(float v) {   // reduce over 16-lane half of wave32
  v = fmaxf(v, __shfl_xor(v, 1, 32));
  v = fmaxf(v, __shfl_xor(v, 2, 32));
  v = fmaxf(v, __shfl_xor(v, 4, 32));
  v = fmaxf(v, __shfl_xor(v, 8, 32));
  return v;
}
__device__ __forceinline__ float half_sum(float v) {
  v += __shfl_xor(v, 1, 32);
  v += __shfl_xor(v, 2, 32);
  v += __shfl_xor(v, 4, 32);
  v += __shfl_xor(v, 8, 32);
  return v;
}

// ---------------------------------------------------------------------------
// Generic Y = X @ W^T ; X:(M,K) bf16, W:(N,K) bf16.
// One wave per 16x(TN*16) strip: TN independent accumulators share each A
// fragment (TN independent WMMAs back-to-back per K-step), K-loop software-
// pipelined (depth 2). OUTMODE: 0 = f32, 1 = f32 + bias, 2 = bf16.
// ---------------------------------------------------------------------------
template <int OUTMODE, int TN>
__global__ __launch_bounds__(128) void gemm_xwt(
    const unsigned short* __restrict__ X, const unsigned short* __restrict__ W,
    const float* __restrict__ bias, void* __restrict__ Y, int M, int N, int K) {
  int lane = threadIdx.x & 31;
  int wid  = threadIdx.x >> 5;
  int n0 = (blockIdx.x * 4 + wid) * (TN * 16);
  int m0 = blockIdx.y * 16;
  int r = lane & 15, hi = lane >> 4, c = r;

  const unsigned short* xr = X + (size_t)(m0 + r) * K + hi * 8;
  const unsigned short* wr[TN];
#pragma unroll
  for (int t = 0; t < TN; t++)
    wr[t] = W + (size_t)(n0 + t * 16 + c) * K + hi * 8;

  v8f zero = {};
  v8f acc[TN];
#pragma unroll
  for (int t = 0; t < TN; t++) acc[t] = zero;

  // pipeline prologue
  v16bf a_cur = ld_frag2(xr, xr + 16);
  v16bf b_cur[TN];
#pragma unroll
  for (int t = 0; t < TN; t++) b_cur[t] = ld_frag2(wr[t], wr[t] + 16);

#pragma unroll 2
  for (int k0 = 0; k0 + 32 < K; k0 += 32) {
    int k1 = k0 + 32;
    if ((k1 & 255) == 0) __builtin_prefetch(xr + k1 + 256, 0, 1);   // global_prefetch_b8
    v16bf a_nxt = ld_frag2(xr + k1, xr + k1 + 16);
    v16bf b_nxt[TN];
#pragma unroll
    for (int t = 0; t < TN; t++) b_nxt[t] = ld_frag2(wr[t] + k1, wr[t] + k1 + 16);
#pragma unroll
    for (int t = 0; t < TN; t++) acc[t] = wmma_bf16(a_cur, b_cur[t], acc[t]);
    a_cur = a_nxt;
#pragma unroll
    for (int t = 0; t < TN; t++) b_cur[t] = b_nxt[t];
  }
#pragma unroll
  for (int t = 0; t < TN; t++) acc[t] = wmma_bf16(a_cur, b_cur[t], acc[t]);

  int row0 = m0 + hi * 8;                   // D layout: VGPR i -> row hi*8+i, col c
  if (OUTMODE == 2) {
    unsigned short* Yb = (unsigned short*)Y;
#pragma unroll
    for (int t = 0; t < TN; t++)
#pragma unroll
      for (int i = 0; i < 8; i++)
        Yb[(size_t)(row0 + i) * N + n0 + t * 16 + c] = f2bf(acc[t][i]);
  } else {
    float* Yf = (float*)Y;
#pragma unroll
    for (int t = 0; t < TN; t++) {
      float bb = 0.0f;
      if (OUTMODE == 1) bb = bias[n0 + t * 16 + c];
#pragma unroll
      for (int i = 0; i < 8; i++)
        Yf[(size_t)(row0 + i) * N + n0 + t * 16 + c] = acc[t][i] + bb;
    }
  }
}

// ---------------------------------------------------------------------------
// Elementwise helpers
// ---------------------------------------------------------------------------
__global__ void cast_f32_bf16_kernel(const float* __restrict__ in,
                                     unsigned short* __restrict__ out, int n) {
  int i = blockIdx.x * blockDim.x + threadIdx.x;
  if (i < n) out[i] = f2bf(in[i]);
}

__global__ void segids_kernel(const int* __restrict__ sp, int* __restrict__ seg) {
  int b = blockIdx.x * blockDim.x + threadIdx.x;
  if (b >= B_) return;
  int run = 0;
  for (int s = 0; s < S_; s++) {
    run += (sp[b * S_ + s] == 0) ? 1 : 0;
    seg[b * S_ + s] = run;
  }
}

// RoPE (first 128 dims rotated, last 128 pass) + cast to bf16.
// in: (B,S,nh,256) f32 ; out: rows strided by out_bstride per batch.
__global__ void rope_cast_kernel(const float* __restrict__ in, const int* __restrict__ segpos,
                                 unsigned short* __restrict__ out, int nh, long out_bstride) {
  long idx = (long)blockIdx.x * blockDim.x + threadIdx.x;
  long total = (long)B_ * S_ * nh * 128;
  if (idx >= total) return;
  int j = (int)(idx & 127);
  long t = idx >> 7;
  int h = (int)(t % nh); t /= nh;
  int s = (int)(t % S_);
  int b = (int)(t / S_);
  const float* row = in + (((size_t)(b * S_ + s)) * nh + h) * HD_;
  unsigned short* orow = out + (size_t)b * out_bstride + ((size_t)s * nh + h) * HD_;
  float pos = (float)segpos[b * S_ + s];
  if (j < 64) {
    float inv_freq = __powf(10000.0f, (float)j * (1.0f / 64.0f));  // 10000^(2f/128)
    float ang = pos / inv_freq;
    float sn = __sinf(ang), cs = __cosf(ang);
    float x1 = row[j], x2 = row[j + 64];
    orow[j]      = f2bf(x1 * cs - x2 * sn);
    orow[j + 64] = f2bf(x2 * cs + x1 * sn);
  } else {
    int d = j + 64;                 // 128..191 and 192..255 pass-through
    orow[d]      = f2bf(row[d]);
    orow[d + 64] = f2bf(row[d + 64]);
  }
}

// V (B,S,HD) bf16  ->  Vt (B,HD,SP_) bf16 so P@V B-operand loads are contiguous
__global__ void transpose_v_kernel(const unsigned short* __restrict__ v,
                                   unsigned short* __restrict__ vt) {
  int idx = blockIdx.x * blockDim.x + threadIdx.x;
  if (idx >= B_ * S_ * HD_) return;
  int d = idx % HD_;
  int t = idx / HD_;
  int s = t % S_;
  int b = t / S_;
  vt[((size_t)b * HD_ + d) * SP_ + s] = v[((size_t)(b * S_ + s)) * HD_ + d];
}

// ---------------------------------------------------------------------------
// Flash attention: one wave per (16-query tile, head). Online softmax,
// logits and P@V via WMMA. Next-block K fragments and first V fragments are
// issued before the softmax VALU phase so their latency hides under it;
// P transposed D-layout -> A-layout through LDS.
// ---------------------------------------------------------------------------
#define ATT_WAVES 4
__global__ __launch_bounds__(32 * ATT_WAVES) void attn_kernel(
    const unsigned short* __restrict__ qbf,   // (B,S,10,256) bf16, RoPE'd
    const unsigned short* __restrict__ kbf,   // (B,SP_,256) bf16, RoPE'd
    const unsigned short* __restrict__ vtbf,  // (B,256,SP_) bf16
    const int* __restrict__ segids,           // (B,S)
    unsigned short* __restrict__ attnbf) {    // (B,S,2560) bf16
  __shared__ unsigned short lds_p[ATT_WAVES][16 * 32];

  int lane = threadIdx.x & 31;
  int wid  = threadIdx.x >> 5;
  int qt = blockIdx.x * ATT_WAVES + wid;
  int h  = blockIdx.y;
  int b  = blockIdx.z;
  int q0 = qt * 16;
  int r = lane & 15, hi = lane >> 4, c = lane & 15;

  const unsigned short* qb = qbf + ((size_t)(b * S_ + q0)) * WIDTH_ + h * HD_;
  const unsigned short* kb = kbf + (size_t)b * SP_ * HD_;
  const unsigned short* vt = vtbf + (size_t)b * HD_ * SP_;
  const int* seg = segids + b * S_;
  unsigned short* lp = lds_p[wid];

  v16bf qfrag[8];                               // 16x256 Q tile, 8 K-chunks of 32
#pragma unroll
  for (int ch = 0; ch < 8; ch++) qfrag[ch] = frag_a(qb, WIDTH_, ch * 32, lane);

  v8f acc[16];                                  // 16 output chunks of 16 dims
  v8f zero = {};
#pragma unroll
  for (int t = 0; t < 16; t++) acc[t] = zero;

  float mrow[8], lrow[8];
  int qp[8], qsg[8];
#pragma unroll
  for (int i = 0; i < 8; i++) {
    mrow[i] = -1e30f;
    lrow[i] = 0.0f;
    qp[i] = q0 + hi * 8 + i;
    qsg[i] = seg[qp[i]];
  }

  const float scale = 0.0625f;                  // 256^-0.5
  int kstart = q0 - WINDOW_;
  if (kstart < 0) kstart = 0;
  kstart &= ~31;
  int kend = q0 + 16;

  // cross-block pipeline: chunk-0 K fragments of the first block
  v16bf bk0c = frag_b(kb, HD_, kstart,      0, lane);
  v16bf bk1c = frag_b(kb, HD_, kstart + 16, 0, lane);

  for (int key0 = kstart; key0 < kend; key0 += 32) {
    // ---- logits S = Q @ K^T for 32 keys; pipelined fragment loads ----
    v8f s0 = zero, s1 = zero;
#pragma unroll
    for (int ch = 0; ch < 8; ch++) {
      v16bf a0 = bk0c, a1 = bk1c;
      if (ch < 7) {
        bk0c = frag_b(kb, HD_, key0,      (ch + 1) * 32, lane);
        bk1c = frag_b(kb, HD_, key0 + 16, (ch + 1) * 32, lane);
      }
      s0 = wmma_bf16(qfrag[ch], a0, s0);
      s1 = wmma_bf16(qfrag[ch], a1, s1);
    }

    // issue next-block K chunk-0 loads and this block's first V loads now,
    // so they complete under the softmax VALU phase below
    bool more = (key0 + 32 < kend);
    if (more) {
      bk0c = frag_b(kb, HD_, key0 + 32, 0, lane);
      bk1c = frag_b(kb, HD_, key0 + 48, 0, lane);
    }
    v16bf bv0 = frag_b(vt, SP_, 0,  key0, lane);
    v16bf bv1 = frag_b(vt, SP_, 16, key0, lane);

    // ---- mask + online softmax update ----
    int kpa = key0 + c, kpb = key0 + 16 + c;
    int ksa = (kpa < S_) ? seg[kpa] : -1;
    int ksb = (kpb < S_) ? seg[kpb] : -1;
    float alpha[8];
#pragma unroll
    for (int i = 0; i < 8; i++) {
      int q = qp[i];
      bool va = (kpa <= q) && (q <= kpa + WINDOW_) && (ksa == qsg[i]);
      bool vb = (kpb <= q) && (q <= kpb + WINDOW_) && (ksb == qsg[i]);
      float t0 = va ? s0[i] * scale : -1e30f;
      float t1 = vb ? s1[i] * scale : -1e30f;
      float bm = half_max(fmaxf(t0, t1));
      float mn = fmaxf(mrow[i], bm);
      float al = __expf(mrow[i] - mn);
      float p0 = __expf(t0 - mn);
      float p1 = __expf(t1 - mn);
      float rs = half_sum(p0 + p1);
      lrow[i] = lrow[i] * al + rs;
      mrow[i] = mn;
      alpha[i] = al;
      // stash P in LDS in D-layout (row = hi*8+i)
      lp[(i + hi * 8) * 32 + c]      = f2bf(p0);
      lp[(i + hi * 8) * 32 + 16 + c] = f2bf(p1);
    }
#pragma unroll
    for (int t = 0; t < 16; t++)
#pragma unroll
      for (int i = 0; i < 8; i++) acc[t][i] *= alpha[i];

    // ---- re-read P as a 16x32 A-fragment (per-wave LDS ops are in-order) ----
    v16bf pf = ld_frag2(lp + r * 32 + hi * 8, lp + r * 32 + 16 + hi * 8);

    // ---- O += P @ V over all 16 head-dim chunks; depth-3 pipelined V loads ----
#pragma unroll
    for (int t = 0; t < 16; t++) {
      v16bf bv = bv0;
      bv0 = bv1;
      if (t < 14) bv1 = frag_b(vt, SP_, (t + 2) * 16, key0, lane);
      acc[t] = wmma_bf16(pf, bv, acc[t]);
    }
  }

  float rl[8];
#pragma unroll
  for (int i = 0; i < 8; i++) rl[i] = 1.0f / lrow[i];

  unsigned short* ob = attnbf + ((size_t)(b * S_ + q0)) * WIDTH_ + h * HD_;
#pragma unroll
  for (int t = 0; t < 16; t++)
#pragma unroll
    for (int i = 0; i < 8; i++)
      ob[(size_t)(hi * 8 + i) * WIDTH_ + t * 16 + c] = f2bf(acc[t][i] * rl[i]);
}

// ---------------------------------------------------------------------------
extern "C" void kernel_launch(void* const* d_in, const int* in_sizes, int n_in,
                              void* d_out, int out_size, void* d_ws, size_t ws_size,
                              hipStream_t stream) {
  (void)in_sizes; (void)n_in; (void)out_size; (void)ws_size;
  const float* x    = (const float*)d_in[0];
  const int*   spos = (const int*)d_in[1];
  const float* Wq   = (const float*)d_in[2];
  const float* Wk   = (const float*)d_in[3];
  const float* Wv   = (const float*)d_in[4];
  const float* Wf   = (const float*)d_in[5];
  const float* bfv  = (const float*)d_in[6];
  float* out = (float*)d_out;

  char* w = (char*)d_ws;
  auto take = [&](size_t bytes) { char* p = w; w += (bytes + 255) & ~(size_t)255; return p; };

  const int M = B_ * S_;
  unsigned short* xbf  = (unsigned short*)take((size_t)M * WIDTH_ * 2);
  unsigned short* wqb  = (unsigned short*)take((size_t)WIDTH_ * WIDTH_ * 2);
  unsigned short* wkb  = (unsigned short*)take((size_t)HD_ * WIDTH_ * 2);
  unsigned short* wvb  = (unsigned short*)take((size_t)HD_ * WIDTH_ * 2);
  unsigned short* wfb  = (unsigned short*)take((size_t)WIDTH_ * WIDTH_ * 2);
  float*          qf   = (float*)take((size_t)M * WIDTH_ * 4);
  float*          kf   = (float*)take((size_t)M * HD_ * 4);
  unsigned short* qbf  = (unsigned short*)take((size_t)M * WIDTH_ * 2);
  unsigned short* kbf  = (unsigned short*)take((size_t)B_ * SP_ * HD_ * 2);
  unsigned short* vbf  = (unsigned short*)take((size_t)M * HD_ * 2);
  unsigned short* vtbf = (unsigned short*)take((size_t)B_ * HD_ * SP_ * 2);
  unsigned short* abf  = (unsigned short*)take((size_t)M * WIDTH_ * 2);
  int*            segi = (int*)take((size_t)M * 4);

  auto cdiv = [](int a, int b) { return (a + b - 1) / b; };

  // bf16 demotion of activations + weights
  cast_f32_bf16_kernel<<<cdiv(M * WIDTH_, 256), 256, 0, stream>>>(x, xbf, M * WIDTH_);
  cast_f32_bf16_kernel<<<cdiv(WIDTH_ * WIDTH_, 256), 256, 0, stream>>>(Wq, wqb, WIDTH_ * WIDTH_);
  cast_f32_bf16_kernel<<<cdiv(HD_ * WIDTH_, 256), 256, 0, stream>>>(Wk, wkb, HD_ * WIDTH_);
  cast_f32_bf16_kernel<<<cdiv(HD_ * WIDTH_, 256), 256, 0, stream>>>(Wv, wvb, HD_ * WIDTH_);
  cast_f32_bf16_kernel<<<cdiv(WIDTH_ * WIDTH_, 256), 256, 0, stream>>>(Wf, wfb, WIDTH_ * WIDTH_);
  segids_kernel<<<1, 32, 0, stream>>>(spos, segi);

  // projections (WMMA bf16 GEMMs)
  // N=2560: 16x128 strip per wave (TN=8) -> grid.x = 2560/(8*16*4) = 5
  // N=256 : 16x64  strip per wave (TN=4) -> grid.x = 256/(4*16*4)  = 1
  gemm_xwt<0, 8><<<dim3(WIDTH_ / 512, M / 16), 128, 0, stream>>>(xbf, wqb, nullptr, qf, M, WIDTH_, WIDTH_);
  gemm_xwt<0, 4><<<dim3(HD_ / 256,   M / 16), 128, 0, stream>>>(xbf, wkb, nullptr, kf, M, HD_, WIDTH_);
  gemm_xwt<2, 4><<<dim3(HD_ / 256,   M / 16), 128, 0, stream>>>(xbf, wvb, nullptr, vbf, M, HD_, WIDTH_);

  // RoPE + demote to bf16
  rope_cast_kernel<<<cdiv(B_ * S_ * NH_ * 128, 256), 256, 0, stream>>>(qf, spos, qbf, NH_, (long)S_ * WIDTH_);
  rope_cast_kernel<<<cdiv(B_ * S_ * 1 * 128, 256), 256, 0, stream>>>(kf, spos, kbf, 1, (long)SP_ * HD_);

  // V^T for contiguous P@V operand loads
  transpose_v_kernel<<<cdiv(B_ * S_ * HD_, 256), 256, 0, stream>>>(vbf, vtbf);

  // windowed causal flash attention
  attn_kernel<<<dim3(S_ / 16 / ATT_WAVES, NH_, B_), 32 * ATT_WAVES, 0, stream>>>(qbf, kbf, vtbf, segi, abf);

  // output projection + bias (fp32 out)
  gemm_xwt<1, 8><<<dim3(WIDTH_ / 512, M / 16), 128, 0, stream>>>(abf, wfb, bfv, out, M, WIDTH_, WIDTH_);
}
